// AttentionBlock_58695023067328
// MI455X (gfx1250) — matrix operand order repaired and checked by reading, not compiled, not run on hardware
//
#include <hip/hip_runtime.h>
#include <hip/hip_bf16.h>

typedef _Float16 v16h __attribute__((ext_vector_type(16)));
typedef _Float16 v8h  __attribute__((ext_vector_type(8)));
typedef float    v8f  __attribute__((ext_vector_type(8)));
typedef int      v4i  __attribute__((ext_vector_type(4)));

union V16U { v16h v; v8h h[2]; };

#if __has_builtin(__builtin_amdgcn_global_load_async_to_lds_b128) && \
    __has_builtin(__builtin_amdgcn_s_wait_asynccnt)
#define USE_ASYNC_LDS 1
#else
#define USE_ASYNC_LDS 0
#endif

__device__ __forceinline__ v8f zero8() {
    v8f z;
#pragma unroll
    for (int i = 0; i < 8; ++i) z[i] = 0.0f;
    return z;
}

// ---------------------------------------------------------------------------
// Kernel 0: f32 -> f16 conversion (weights)
// ---------------------------------------------------------------------------
__global__ void f32_to_f16_kernel(const float* __restrict__ src,
                                  _Float16* __restrict__ dst, int n) {
    int i = blockIdx.x * blockDim.x + threadIdx.x;
    if (i < n) dst[i] = (_Float16)src[i];
}

// ---------------------------------------------------------------------------
// Kernel 1: GroupNorm(32, 512) -> f16, transposed to (b, s, c) layout
// one block per (b, group): 16 channels x 1024 spatial
// ---------------------------------------------------------------------------
__global__ void groupnorm_kernel(const float* __restrict__ x,
                                 const float* __restrict__ gamma,
                                 const float* __restrict__ beta,
                                 _Float16* __restrict__ xnT) {
    const int bg  = blockIdx.x;       // 0..255
    const int b   = bg >> 5;
    const int g   = bg & 31;
    const int tid = threadIdx.x;      // 256 threads
    const float* xb = x + (size_t)(b * 512 + g * 16) * 1024;

    float s = 0.0f, s2 = 0.0f;
#pragma unroll
    for (int i = 0; i < 64; ++i) {
        float v = xb[tid + i * 256];
        s += v; s2 += v * v;
    }
    __shared__ float red[256], red2[256];
    red[tid] = s; red2[tid] = s2;
    __syncthreads();
    for (int off = 128; off > 0; off >>= 1) {
        if (tid < off) { red[tid] += red[tid + off]; red2[tid] += red2[tid + off]; }
        __syncthreads();
    }
    const float inv_n = 1.0f / 16384.0f;
    const float mu  = red[0] * inv_n;
    const float var = red2[0] * inv_n - mu * mu;
    const float rs  = rsqrtf(var + 1e-5f);

    for (int i = tid; i < 16384; i += 256) {
        int cl = i >> 10;         // channel in group
        int sp = i & 1023;        // spatial
        int c  = g * 16 + cl;
        float v = (xb[i] - mu) * rs * gamma[c] + beta[c];
        xnT[((size_t)b * 1024 + sp) * 512 + c] = (_Float16)v;
    }
}

// ---------------------------------------------------------------------------
// Kernel 2: QKV GEMM  (3C=1536, C=512) x (C=512, S=1024 per batch)
// each wave: 16x64 output tile via 4 WMMA accumulators, K loop 512/32
// epilogue: +bias, scale q/k by D^-0.25, scatter to (b,h,s,d) f16
// ---------------------------------------------------------------------------
__global__ void __launch_bounds__(256) qkv_gemm_kernel(
        const _Float16* __restrict__ wq, const _Float16* __restrict__ xnT,
        const float* __restrict__ b_qkv,
        _Float16* __restrict__ q16, _Float16* __restrict__ k16,
        _Float16* __restrict__ v16) {
    const int lane = threadIdx.x & 31;
    const int wv   = threadIdx.x >> 5;
    const int unit = blockIdx.x * 8 + wv;       // 0..12287 = 8 * 96 * 16
    const int b    = unit / (96 * 16);
    const int rem  = unit % (96 * 16);
    const int m0   = (rem >> 4) * 16;
    const int n0   = (rem & 15) * 64;
    const int half = lane >> 4, l15 = lane & 15;

    v8f acc[4];
#pragma unroll
    for (int j = 0; j < 4; ++j) acc[j] = zero8();

    const _Float16* arow  = wq + (size_t)(m0 + l15) * 512;
    const _Float16* bbase = xnT + (size_t)b * 1024 * 512;

    for (int kk = 0; kk < 512; kk += 32) {
        V16U a;
        a.h[0] = *(const v8h*)(arow + kk + half * 16);
        a.h[1] = *(const v8h*)(arow + kk + half * 16 + 8);
#pragma unroll
        for (int j = 0; j < 4; ++j) {
            const _Float16* bp = bbase + (size_t)(n0 + j * 16 + l15) * 512 + kk + half * 16;
            V16U bf;
            bf.h[0] = *(const v8h*)(bp);
            bf.h[1] = *(const v8h*)(bp + 8);
            acc[j] = __builtin_amdgcn_wmma_f32_16x16x32_f16(
                false, a.v, false, bf.v, (short)0, acc[j], false, false);
        }
    }

    const int t     = m0 / 512;                 // 0=q 1=k 2=v
    const int hh    = (m0 % 512) / 64;          // head
    const int dbase = (m0 & 63) + half * 8;     // d offset of this lane's rows
    const float mult = (t < 2) ? 0.35355339059327373f : 1.0f;
    float bias[8];
#pragma unroll
    for (int r = 0; r < 8; ++r) bias[r] = b_qkv[m0 + half * 8 + r];
    _Float16* sel = (t == 0) ? q16 : ((t == 1) ? k16 : v16);
#pragma unroll
    for (int j = 0; j < 4; ++j) {
        const int s = n0 + j * 16 + l15;
        v8h pk;
#pragma unroll
        for (int r = 0; r < 8; ++r)
            pk[r] = (_Float16)((acc[j][r] + bias[r]) * mult);
        *(v8h*)(sel + ((size_t)(b * 8 + hh) * 1024 + s) * 64 + dbase) = pk;
    }
}

// ---------------------------------------------------------------------------
// Kernel 3: flash attention per (b,h); block = 128 query rows (8 waves x 16),
// K/V key blocks of 32 staged in LDS (K via async global->LDS copy when the
// toolchain exposes it; V transposed), online softmax, QK^T and PV both on
// v_wmma_f32_16x16x32_f16.  Output to (b, s, c) f16.
// ---------------------------------------------------------------------------
__global__ void __launch_bounds__(256) attn_kernel(
        const _Float16* __restrict__ q16, const _Float16* __restrict__ k16,
        const _Float16* __restrict__ v16, _Float16* __restrict__ attnT) {
    __shared__ _Float16 Kt[32][64];     // K rows (s', d)
    __shared__ _Float16 VtT[64][32];    // V transposed (d, s')
    __shared__ _Float16 Pb[8][16][32];  // per-wave P relayout buffer

    const int lane  = threadIdx.x & 31;
    const int wv    = threadIdx.x >> 5;
    const int bh    = blockIdx.x >> 3;  // 0..63
    const int qt    = blockIdx.x & 7;
    const int half  = lane >> 4, l15 = lane & 15;
    const int qrow0 = qt * 128 + wv * 16;
    const size_t base = (size_t)bh * 1024 * 64;

    // Q A-fragments for K-steps d=0..31 and d=32..63
    V16U qa[2];
    {
        const _Float16* qrow = q16 + base + (size_t)(qrow0 + l15) * 64;
#pragma unroll
        for (int kd = 0; kd < 2; ++kd) {
            qa[kd].h[0] = *(const v8h*)(qrow + kd * 32 + half * 16);
            qa[kd].h[1] = *(const v8h*)(qrow + kd * 32 + half * 16 + 8);
        }
    }

    v8f acc[4];
#pragma unroll
    for (int dt = 0; dt < 4; ++dt) acc[dt] = zero8();
    float mmax[8], lsum[8];
#pragma unroll
    for (int r = 0; r < 8; ++r) { mmax[r] = -1e30f; lsum[r] = 0.0f; }

    const int tid   = threadIdx.x;
    const int ldrow = tid >> 3;        // 0..31 key row
    const int lddo  = (tid & 7) * 8;   // d offset

    for (int kb = 0; kb < 1024; kb += 32) {
        __syncthreads();
        {   // cooperative K/V tile load; V transposed into LDS
#if USE_ASYNC_LDS
            __builtin_amdgcn_global_load_async_to_lds_b128(
                (v4i __attribute__((address_space(1)))*)
                    (k16 + base + (size_t)(kb + ldrow) * 64 + lddo),
                (v4i __attribute__((address_space(3)))*)(&Kt[ldrow][lddo]),
                0, 0);
#else
            v8h kv = *(const v8h*)(k16 + base + (size_t)(kb + ldrow) * 64 + lddo);
            *(v8h*)(&Kt[ldrow][lddo]) = kv;
#endif
            v8h vl = *(const v8h*)(v16 + base + (size_t)(kb + ldrow) * 64 + lddo);
#pragma unroll
            for (int i = 0; i < 8; ++i) VtT[lddo + i][ldrow] = vl[i];
#if USE_ASYNC_LDS
            __builtin_amdgcn_s_wait_asynccnt(0);
#endif
        }
        __syncthreads();

        // logits: 16 queries x 32 keys (two 16x16 n-tiles, 2 K-steps over d)
        v8f lf[2];
#pragma unroll
        for (int j = 0; j < 2; ++j) {
            lf[j] = zero8();
#pragma unroll
            for (int kd = 0; kd < 2; ++kd) {
                V16U bf;
                bf.h[0] = *(const v8h*)(&Kt[j * 16 + l15][kd * 32 + half * 16]);
                bf.h[1] = *(const v8h*)(&Kt[j * 16 + l15][kd * 32 + half * 16 + 8]);
                lf[j] = __builtin_amdgcn_wmma_f32_16x16x32_f16(
                    false, qa[kd].v, false, bf.v, (short)0, lf[j], false, false);
            }
        }

        // online softmax (row = 8*half + r; cols spread over 16-lane group)
        float corr[8];
#pragma unroll
        for (int r = 0; r < 8; ++r) {
            float v0 = lf[0][r], v1 = lf[1][r];
            float tmax = fmaxf(v0, v1);
#pragma unroll
            for (int mk = 1; mk <= 8; mk <<= 1)
                tmax = fmaxf(tmax, __shfl_xor(tmax, mk, 32));
            float mnew = fmaxf(mmax[r], tmax);
            corr[r] = __expf(mmax[r] - mnew);
            float p0 = __expf(v0 - mnew);
            float p1 = __expf(v1 - mnew);
            float ps = p0 + p1;
#pragma unroll
            for (int mk = 1; mk <= 8; mk <<= 1)
                ps += __shfl_xor(ps, mk, 32);
            lsum[r] = lsum[r] * corr[r] + ps;
            mmax[r] = mnew;
            Pb[wv][half * 8 + r][l15]      = (_Float16)p0;
            Pb[wv][half * 8 + r][16 + l15] = (_Float16)p1;
        }
#pragma unroll
        for (int dt = 0; dt < 4; ++dt)
#pragma unroll
            for (int r = 0; r < 8; ++r)
                acc[dt][r] *= corr[r];

        // P relayout (C-layout -> A-fragment) through per-wave LDS
        V16U pa;
        pa.h[0] = *(const v8h*)(&Pb[wv][l15][half * 16]);
        pa.h[1] = *(const v8h*)(&Pb[wv][l15][half * 16 + 8]);

#pragma unroll
        for (int dt = 0; dt < 4; ++dt) {
            V16U vb;
            vb.h[0] = *(const v8h*)(&VtT[dt * 16 + l15][half * 16]);
            vb.h[1] = *(const v8h*)(&VtT[dt * 16 + l15][half * 16 + 8]);
            acc[dt] = __builtin_amdgcn_wmma_f32_16x16x32_f16(
                false, pa.v, false, vb.v, (short)0, acc[dt], false, false);
        }
    }

    // epilogue: normalize, store transposed to (b, s, c=h*64+d) for proj GEMM
    const int b = bh >> 3, h = bh & 7;
    float inv[8];
#pragma unroll
    for (int r = 0; r < 8; ++r) inv[r] = 1.0f / lsum[r];
#pragma unroll
    for (int dt = 0; dt < 4; ++dt) {
        const int d = dt * 16 + l15;
#pragma unroll
        for (int r = 0; r < 8; ++r) {
            const int sq = qrow0 + half * 8 + r;
            attnT[((size_t)b * 1024 + sq) * 512 + h * 64 + d] =
                (_Float16)(acc[dt][r] * inv[r]);
        }
    }
}

// ---------------------------------------------------------------------------
// Kernel 4: proj GEMM (512x512) x (512, S) + bias + residual -> f32 out
// ---------------------------------------------------------------------------
__global__ void __launch_bounds__(256) proj_gemm_kernel(
        const _Float16* __restrict__ wp, const _Float16* __restrict__ attnT,
        const float* __restrict__ b_proj, const float* __restrict__ x,
        float* __restrict__ out) {
    const int lane = threadIdx.x & 31;
    const int wv   = threadIdx.x >> 5;
    const int unit = blockIdx.x * 8 + wv;     // 0..4095 = 8 * 32 * 16
    const int b    = unit / (32 * 16);
    const int rem  = unit % (32 * 16);
    const int m0   = (rem >> 4) * 16;
    const int n0   = (rem & 15) * 64;
    const int half = lane >> 4, l15 = lane & 15;

    v8f acc[4];
#pragma unroll
    for (int j = 0; j < 4; ++j) acc[j] = zero8();

    const _Float16* arow  = wp + (size_t)(m0 + l15) * 512;
    const _Float16* bbase = attnT + (size_t)b * 1024 * 512;

    for (int kk = 0; kk < 512; kk += 32) {
        V16U a;
        a.h[0] = *(const v8h*)(arow + kk + half * 16);
        a.h[1] = *(const v8h*)(arow + kk + half * 16 + 8);
#pragma unroll
        for (int j = 0; j < 4; ++j) {
            const _Float16* bp = bbase + (size_t)(n0 + j * 16 + l15) * 512 + kk + half * 16;
            V16U bf;
            bf.h[0] = *(const v8h*)(bp);
            bf.h[1] = *(const v8h*)(bp + 8);
            acc[j] = __builtin_amdgcn_wmma_f32_16x16x32_f16(
                false, a.v, false, bf.v, (short)0, acc[j], false, false);
        }
    }

#pragma unroll
    for (int j = 0; j < 4; ++j) {
        const int s = n0 + j * 16 + l15;
#pragma unroll
        for (int r = 0; r < 8; ++r) {
            const int o = m0 + half * 8 + r;
            const size_t idx = ((size_t)(b * 512 + o)) * 1024 + s;
            out[idx] = acc[j][r] + b_proj[o] + x[idx];
        }
    }
}

// ---------------------------------------------------------------------------
// launch
// ---------------------------------------------------------------------------
extern "C" void kernel_launch(void* const* d_in, const int* in_sizes, int n_in,
                              void* d_out, int out_size, void* d_ws, size_t ws_size,
                              hipStream_t stream) {
    (void)in_sizes; (void)n_in; (void)out_size; (void)ws_size;
    const float* x      = (const float*)d_in[0];
    const float* gamma  = (const float*)d_in[1];
    const float* beta   = (const float*)d_in[2];
    const float* w_qkv  = (const float*)d_in[3];
    const float* b_qkv  = (const float*)d_in[4];
    const float* w_proj = (const float*)d_in[5];
    const float* b_proj = (const float*)d_in[6];
    float* out = (float*)d_out;

    char* ws = (char*)d_ws;
    _Float16* wq16  = (_Float16*)(ws + 0);          // 1536*512*2 = 1572864
    _Float16* wp16  = (_Float16*)(ws + 1572864);    // 512*512*2  = 524288
    _Float16* xnT   = (_Float16*)(ws + 2097152);    // 8*1024*512*2 = 8388608
    _Float16* q16   = (_Float16*)(ws + 10485760);   // 8*8*1024*64*2 = 8388608
    _Float16* k16   = (_Float16*)(ws + 18874368);
    _Float16* v16   = (_Float16*)(ws + 27262976);
    _Float16* attnT = (_Float16*)(ws + 35651584);   // ends at 44040192

    f32_to_f16_kernel<<<(786432 + 255) / 256, 256, 0, stream>>>(w_qkv, wq16, 786432);
    f32_to_f16_kernel<<<(262144 + 255) / 256, 256, 0, stream>>>(w_proj, wp16, 262144);
    groupnorm_kernel<<<256, 256, 0, stream>>>(x, gamma, beta, xnT);
    qkv_gemm_kernel<<<1536, 256, 0, stream>>>(wq16, xnT, b_qkv, q16, k16, v16);
    attn_kernel<<<512, 256, 0, stream>>>(q16, k16, v16, attnT);
    proj_gemm_kernel<<<512, 256, 0, stream>>>(wp16, attnT, b_proj, x, out);
}
